// GCN_BiLSTM_Classifier_49761491091774
// MI455X (gfx1250) — matrix operand order repaired and checked by reading, not compiled, not run on hardware
//
#include <hip/hip_runtime.h>
#include <hip/hip_bf16.h>

// ---------------------------------------------------------------------------
// CDNA5 (gfx1250) implementation: GCN (2x A-GEMM) -> gather -> BiLSTM -> FC
// All heavy math on v_wmma_f32_16x16x32_bf16 (fp32 accumulate).
// GEMM waves compute 32x32 output (2x2 WMMA tiles) to halve fragment traffic.
// ---------------------------------------------------------------------------

typedef __attribute__((ext_vector_type(16))) __bf16 v16bf;
typedef __attribute__((ext_vector_type(8)))  float  v8f;

union Frag { v16bf v; uint4 q[2]; };

struct bf16x4 { __bf16 x, y, z, w; };

static __device__ __forceinline__ v8f wmma_bf16(const Frag& a, const Frag& b, v8f c) {
  return __builtin_amdgcn_wmma_f32_16x16x32_bf16(
      false, a.v, false, b.v, (short)0, c, false, false);
}

// ---------------- fp32 -> bf16 row-major conversion (vectorized) -----------
__global__ void cvt_f32_to_bf16(const float* __restrict__ src,
                                __bf16* __restrict__ dst, long n) {
  long i = ((long)blockIdx.x * blockDim.x + threadIdx.x) * 4;
  long stride = (long)gridDim.x * blockDim.x * 4;
  for (; i < n; i += stride) {
    float4 v = *(const float4*)(src + i);
    bf16x4 o{(__bf16)v.x, (__bf16)v.y, (__bf16)v.z, (__bf16)v.w};
    *(bf16x4*)(dst + i) = o;
  }
}

// ---------------- pack fp32 B[KxN] (optionally stored [NxK]) into ----------
// ---------------- bf16 WMMA B-fragment tiles: dst[(nt*KT+kt)*512+L*16+j] ---
// lane L: col n = nt*16 + (L&15); k = kt*32 + (L>>4)*16 + j  (j = 0..15)
__global__ void pack_b_frag(const float* __restrict__ src,
                            __bf16* __restrict__ dst,
                            int K, int N, int transposed) {
  int tid = blockIdx.x * blockDim.x + threadIdx.x;
  int total = K * N;
  if (tid >= total) return;
  int j = tid & 15;
  int L = (tid >> 4) & 31;
  int t = tid >> 9;            // tile index = nt*KT + kt
  int KT = K >> 5;
  int kt = t % KT;
  int nt = t / KT;
  int n = nt * 16 + (L & 15);
  int k = kt * 32 + (L >> 4) * 16 + j;
  float v = transposed ? src[(long)n * K + k] : src[(long)k * N + n];
  dst[tid] = (__bf16)v;
}

// ---------------- GEMM: C[MxN] = A[MxK](bf16 row-major) * Bpack + bias -----
// block = 256 threads = 8 waves in 2(row)x4(col); wave tile = 32x32 (2x2 WMMA)
// macro tile = 64 x 128.  Requires M%64==0, N%128==0, K%32==0.
__global__ __launch_bounds__(256) void gemm_bf16_wmma(
    const __bf16* __restrict__ A, const __bf16* __restrict__ Bp,
    const float* __restrict__ bias, float* __restrict__ C,
    int M, int N, int K) {
  const int lane = threadIdx.x & 31, wave = threadIdx.x >> 5;
  const int wrow = wave >> 2;                       // 0..1
  const int wcol = wave & 3;                        // 0..3
  const int rowBase = blockIdx.y * 64 + wrow * 32;  // two 16-row tiles
  const int colBase = blockIdx.x * 128 + wcol * 32; // two 16-col tiles
  const int nt0 = colBase >> 4;
  const int llo = lane & 15, lhi = lane >> 4;
  const int KT = K >> 5;

  v8f acc00 = {0.f, 0.f, 0.f, 0.f, 0.f, 0.f, 0.f, 0.f};
  v8f acc01 = acc00, acc10 = acc00, acc11 = acc00;

  const __bf16* arow0 = A + (long)(rowBase + llo) * K;
  const __bf16* arow1 = arow0 + (long)16 * K;
  const __bf16* bbA   = Bp + (long)nt0 * KT * 512 + lane * 16;
  const __bf16* bbB   = bbA + (long)KT * 512;

  for (int kt = 0; kt < KT; kt++) {
    const int kb = kt * 32 + lhi * 16;
    __builtin_prefetch(arow0 + kb + 128, 0, 1);   // global_prefetch_b8
    __builtin_prefetch(arow1 + kb + 128, 0, 1);
    Frag a0, a1, b0, b1;
    a0.q[0] = *(const uint4*)(arow0 + kb);
    a0.q[1] = *(const uint4*)(arow0 + kb + 8);
    a1.q[0] = *(const uint4*)(arow1 + kb);
    a1.q[1] = *(const uint4*)(arow1 + kb + 8);
    const __bf16* bp0 = bbA + kt * 512;
    const __bf16* bp1 = bbB + kt * 512;
    b0.q[0] = *(const uint4*)bp0;  b0.q[1] = *(const uint4*)(bp0 + 8);
    b1.q[0] = *(const uint4*)bp1;  b1.q[1] = *(const uint4*)(bp1 + 8);
    acc00 = wmma_bf16(a0, b0, acc00);
    acc01 = wmma_bf16(a0, b1, acc01);
    acc10 = wmma_bf16(a1, b0, acc10);
    acc11 = wmma_bf16(a1, b1, acc11);
  }

  float bv0 = bias ? bias[colBase + llo] : 0.0f;
  float bv1 = bias ? bias[colBase + 16 + llo] : 0.0f;
#pragma unroll
  for (int e = 0; e < 8; e++) {
    int r0 = rowBase + lhi * 8 + e;
    int r1 = r0 + 16;
    C[(long)r0 * N + colBase + llo]      = acc00[e] + bv0;
    C[(long)r0 * N + colBase + 16 + llo] = acc01[e] + bv1;
    C[(long)r1 * N + colBase + llo]      = acc10[e] + bv0;
    C[(long)r1 * N + colBase + 16 + llo] = acc11[e] + bv1;
  }
}

// ---------------- gather Hn rows (bf16) into fwd/bwd sequences -------------
// seqF[(l*64+b)*128+d] = Hnbf[tok[b*64+l]*128+d]; seqB uses l -> 63-l
__global__ void gather_seqs(const __bf16* __restrict__ Hnbf,
                            const int* __restrict__ tok,
                            __bf16* __restrict__ seqF,
                            __bf16* __restrict__ seqB) {
  int tid = blockIdx.x * blockDim.x + threadIdx.x;   // 2*4096*128 = 2^20
  int d = tid & 127;
  int r = (tid >> 7) & 4095;
  int dir = tid >> 19;
  int l = r >> 6, b = r & 63;
  int ll = dir ? (63 - l) : l;
  int t = tok[b * 64 + ll];
  __bf16 v = Hnbf[(long)t * 128 + d];
  if (dir) seqB[(long)r * 128 + d] = v;
  else     seqF[(long)r * 128 + d] = v;
}

// ---------------- LSTM: one block per direction, 32 waves ------------------
// Wave (rt,jt) owns batch rows rt*16..+15 and j cols jt*16..+15 for ALL four
// gates (N-tiles jt, jt+8, jt+16, jt+24), so i/f/g/o land element-aligned in
// registers; c,h live in registers in C-fragment layout; h republished to LDS
// as bf16 each step for the next h @ Whh^T WMMA.
__global__ __launch_bounds__(1024) void lstm_wmma(
    const float* __restrict__ xpF, const float* __restrict__ xpB,
    const __bf16* __restrict__ WhhFp, const __bf16* __restrict__ WhhBp,
    const int* __restrict__ lengths,
    float* __restrict__ hF, float* __restrict__ hB) {
  const int dir = blockIdx.x;
  const float*  xp   = dir ? xpB   : xpF;
  const __bf16* Whp  = dir ? WhhBp : WhhFp;
  float*        hout = dir ? hB    : hF;

  __shared__ __attribute__((aligned(16))) __bf16 hbf[64 * 128];

  const int tid = threadIdx.x;
  const int lane = tid & 31, wave = tid >> 5;
  const int rt = wave >> 3;     // batch tile 0..3
  const int jt = wave & 7;      // j tile 0..7
  const int llo = lane & 15, lhi = lane >> 4;

  for (int i = tid; i < 64 * 128; i += 1024) hbf[i] = (__bf16)0.0f;
  __syncthreads();

  float c_reg[8], h_reg[8];
  int len_e[8];
#pragma unroll
  for (int e = 0; e < 8; e++) {
    c_reg[e] = 0.f; h_reg[e] = 0.f;
    len_e[e] = lengths[rt * 16 + lhi * 8 + e];
  }

  for (int l = 0; l < 64; l++) {
    v8f acc[4];
#pragma unroll
    for (int g = 0; g < 4; g++) {
#pragma unroll
      for (int e = 0; e < 8; e++) {
        int be = rt * 16 + lhi * 8 + e;
        acc[g][e] = xp[(long)(l * 64 + be) * 512 + g * 128 + jt * 16 + llo];
      }
    }
#pragma unroll
    for (int kt = 0; kt < 4; kt++) {
      Frag a;
      int arow = rt * 16 + llo;
      int kb = kt * 32 + lhi * 16;
      a.q[0] = *(const uint4*)&hbf[arow * 128 + kb];
      a.q[1] = *(const uint4*)&hbf[arow * 128 + kb + 8];
#pragma unroll
      for (int g = 0; g < 4; g++) {
        int nt = g * 8 + jt;
        Frag b;
        const __bf16* bp = Whp + (nt * 4 + kt) * 512 + lane * 16;
        b.q[0] = *(const uint4*)bp;
        b.q[1] = *(const uint4*)(bp + 8);
        acc[g] = wmma_bf16(a, b, acc[g]);
      }
    }
    __syncthreads();   // everyone done reading hbf
    int step = dir ? (63 - l) : l;
#pragma unroll
    for (int e = 0; e < 8; e++) {
      int be = rt * 16 + lhi * 8 + e;
      int j  = jt * 16 + llo;
      float gi = acc[0][e], gf = acc[1][e], gg = acc[2][e], go = acc[3][e];
      float si = 1.f / (1.f + __expf(-gi));
      float sf = 1.f / (1.f + __expf(-gf));
      float so = 1.f / (1.f + __expf(-go));
      float cn = sf * c_reg[e] + si * tanhf(gg);
      float hn = so * tanhf(cn);
      bool m = step < len_e[e];
      c_reg[e] = m ? cn : c_reg[e];
      h_reg[e] = m ? hn : h_reg[e];
      hbf[be * 128 + j] = (__bf16)h_reg[e];
    }
    __syncthreads();   // hbf updated for next step
  }
#pragma unroll
  for (int e = 0; e < 8; e++) {
    int be = rt * 16 + lhi * 8 + e;
    hout[be * 128 + jt * 16 + llo] = h_reg[e];
  }
}

// ---------------- final FC: out[b][c] = [hF,hB] @ Wfc^T + bfc --------------
__global__ void final_fc(const float* __restrict__ hF, const float* __restrict__ hB,
                         const float* __restrict__ Wfc, const float* __restrict__ bfc,
                         float* __restrict__ out) {
  int tid = threadIdx.x;       // 256 = 64 batch * 4 classes
  int b = tid >> 2, cc = tid & 3;
  float s = bfc[cc];
  for (int j = 0; j < 128; j++) {
    s += hF[b * 128 + j] * Wfc[cc * 256 + j];
    s += hB[b * 128 + j] * Wfc[cc * 256 + 128 + j];
  }
  out[b * 4 + cc] = s;
}

// ---------------------------------------------------------------------------
extern "C" void kernel_launch(void* const* d_in, const int* in_sizes, int n_in,
                              void* d_out, int out_size, void* d_ws, size_t ws_size,
                              hipStream_t stream) {
  const float* A     = (const float*)d_in[0];
  const float* X     = (const float*)d_in[1];
  const int*   tok   = (const int*)d_in[2];
  const int*   lens  = (const int*)d_in[3];
  const float* W1    = (const float*)d_in[4];
  const float* b1    = (const float*)d_in[5];
  const float* W2    = (const float*)d_in[6];
  const float* b2    = (const float*)d_in[7];
  const float* Wih_f = (const float*)d_in[8];
  const float* Whh_f = (const float*)d_in[9];
  const float* b_f   = (const float*)d_in[10];
  const float* Wih_b = (const float*)d_in[11];
  const float* Whh_b = (const float*)d_in[12];
  const float* b_b   = (const float*)d_in[13];
  const float* Wfc   = (const float*)d_in[14];
  const float* bfc   = (const float*)d_in[15];
  float* out = (float*)d_out;

  const int N = 8192, DIN = 256, HID = 256, DOUT = 128;

  char* ws = (char*)d_ws;
  size_t off = 0;
  auto alloc = [&](size_t bytes) -> char* {
    char* p = ws + off;
    off = (off + bytes + 255) & ~(size_t)255;
    return p;
  };
  __bf16* Abf   = (__bf16*)alloc((size_t)N * N * 2);        // 128 MB
  __bf16* Xbf   = (__bf16*)alloc((size_t)N * DIN * 2);
  __bf16* W1p   = (__bf16*)alloc((size_t)DIN * HID * 2);
  float*  P1    = (float*) alloc((size_t)N * HID * 4);
  __bf16* P1p   = (__bf16*)alloc((size_t)N * HID * 2);
  float*  H1    = (float*) alloc((size_t)N * HID * 4);
  __bf16* H1bf  = (__bf16*)alloc((size_t)N * HID * 2);
  __bf16* W2p   = (__bf16*)alloc((size_t)HID * DOUT * 2);
  float*  P2    = (float*) alloc((size_t)N * DOUT * 4);
  __bf16* P2p   = (__bf16*)alloc((size_t)N * DOUT * 2);
  float*  Hn    = (float*) alloc((size_t)N * DOUT * 4);
  __bf16* Hnbf  = (__bf16*)alloc((size_t)N * DOUT * 2);
  __bf16* seqF  = (__bf16*)alloc((size_t)4096 * 128 * 2);
  __bf16* seqB  = (__bf16*)alloc((size_t)4096 * 128 * 2);
  __bf16* WihFp = (__bf16*)alloc((size_t)128 * 512 * 2);
  __bf16* WihBp = (__bf16*)alloc((size_t)128 * 512 * 2);
  __bf16* WhhFp = (__bf16*)alloc((size_t)128 * 512 * 2);
  __bf16* WhhBp = (__bf16*)alloc((size_t)128 * 512 * 2);
  float*  xpF   = (float*) alloc((size_t)4096 * 512 * 4);
  float*  xpB   = (float*) alloc((size_t)4096 * 512 * 4);
  float*  hF    = (float*) alloc((size_t)64 * 128 * 4);
  float*  hB    = (float*) alloc((size_t)64 * 128 * 4);

  // 1) precision conversion of streaming operands
  cvt_f32_to_bf16<<<8192, 256, 0, stream>>>(A, Abf, (long)N * N);
  cvt_f32_to_bf16<<<512,  256, 0, stream>>>(X, Xbf, (long)N * DIN);

  // 2) P1 = X @ W1
  pack_b_frag<<<(DIN * HID + 255) / 256, 256, 0, stream>>>(W1, W1p, DIN, HID, 0);
  gemm_bf16_wmma<<<dim3(HID / 128, N / 64), 256, 0, stream>>>(Xbf, W1p, nullptr, P1, N, HID, DIN);

  // 3) H1 = A @ P1 + b1
  pack_b_frag<<<(N * HID + 255) / 256, 256, 0, stream>>>(P1, P1p, N, HID, 0);
  gemm_bf16_wmma<<<dim3(HID / 128, N / 64), 256, 0, stream>>>(Abf, P1p, b1, H1, N, HID, N);

  // 4) P2 = H1 @ W2
  cvt_f32_to_bf16<<<512, 256, 0, stream>>>(H1, H1bf, (long)N * HID);
  pack_b_frag<<<(HID * DOUT + 255) / 256, 256, 0, stream>>>(W2, W2p, HID, DOUT, 0);
  gemm_bf16_wmma<<<dim3(DOUT / 128, N / 64), 256, 0, stream>>>(H1bf, W2p, nullptr, P2, N, DOUT, HID);

  // 5) Hn = A @ P2 + b2
  pack_b_frag<<<(N * DOUT + 255) / 256, 256, 0, stream>>>(P2, P2p, N, DOUT, 0);
  gemm_bf16_wmma<<<dim3(DOUT / 128, N / 64), 256, 0, stream>>>(Abf, P2p, b2, Hn, N, DOUT, N);

  // 6) gather sequences (bf16)
  cvt_f32_to_bf16<<<256, 256, 0, stream>>>(Hn, Hnbf, (long)N * DOUT);
  gather_seqs<<<4096, 256, 0, stream>>>(Hnbf, tok, seqF, seqB);

  // 7) xp = seqs @ Wih^T + b   (M=4096, N=512, K=128)
  pack_b_frag<<<256, 256, 0, stream>>>(Wih_f, WihFp, 128, 512, 1);
  pack_b_frag<<<256, 256, 0, stream>>>(Wih_b, WihBp, 128, 512, 1);
  gemm_bf16_wmma<<<dim3(512 / 128, 4096 / 64), 256, 0, stream>>>(seqF, WihFp, b_f, xpF, 4096, 512, 128);
  gemm_bf16_wmma<<<dim3(512 / 128, 4096 / 64), 256, 0, stream>>>(seqB, WihBp, b_b, xpB, 4096, 512, 128);

  // 8) BiLSTM (one workgroup per direction)
  pack_b_frag<<<256, 256, 0, stream>>>(Whh_f, WhhFp, 128, 512, 1);
  pack_b_frag<<<256, 256, 0, stream>>>(Whh_b, WhhBp, 128, 512, 1);
  lstm_wmma<<<2, 1024, 0, stream>>>(xpF, xpB, WhhFp, WhhBp, lens, hF, hB);

  // 9) classifier head
  final_fc<<<1, 256, 0, stream>>>(hF, hB, Wfc, bfc, out);
}
